// TLSTM_49520972923546
// MI455X (gfx1250) — matrix operand order repaired and checked by reading, not compiled, not run on hardware
//
#include <hip/hip_runtime.h>

// ---------------------------------------------------------------------------
// T-LSTM (2 layers) for MI455X / gfx1250.
//   - bf16 WMMA (v_wmma_f32_16x16x32_bf16), fp32 accumulate & elementwise.
//   - One workgroup per 16-batch-row tile; full S=512 scan inside the WG.
//   - Fused [x_t, h] @ [Wx; Wh] GEMM (K=512, N=1024) + c@Wd GEMM per step.
//   - Weights pre-swizzled into B-fragment layout (contiguous k per lane half); stream from L2
//     every step. The uniform k-step offset is laundered through an SGPR
//     asm so the loads stay global_load_b128 inside the sequence loop
//     (no LICM mass-hoist + scratch spill, no flat-address fallback).
// ---------------------------------------------------------------------------

typedef __attribute__((ext_vector_type(16))) __bf16 v16bf;
typedef __attribute__((ext_vector_type(8)))  float  v8f;

union BF16Frag { v16bf v; uint4 u[2]; unsigned short s[16]; };

__device__ __forceinline__ unsigned short f2bf(float f) {
  unsigned int u = __float_as_uint(f);
  u += 0x7FFFu + ((u >> 16) & 1u);           // round-to-nearest-even
  return (unsigned short)(u >> 16);
}
__device__ __forceinline__ float bf2f(unsigned short h) {
  return __uint_as_float(((unsigned int)h) << 16);
}
__device__ __forceinline__ float sigf(float x) {
  return 1.0f / (1.0f + __expf(-x));
}
// Opaque uniform 32-bit offset (SGPR). asm volatile is never CSE'd/hoisted,
// so addresses derived from it cannot be proven loop-invariant -> weight
// loads stay inside the sequence loop. Pointer provenance is untouched, so
// loads remain addrspace(1) global_load_b128 (not flat).
__device__ __forceinline__ unsigned lau(unsigned x) {
  asm volatile("" : "+s"(x));
  return x;
}

static constexpr int Hdim = 256;
static constexpr int Sdim = 512;
static constexpr int Bdim = 256;
static constexpr int WCAT_ELEMS = 512 * 1024;   // [K=512][N=4H=1024]
static constexpr int WD_ELEMS   = 256 * 256;    // [K=256][N=256]

// B-fragment swizzle (this file's contiguous-k order, not the ISA table order; 16-bit B KxN):
//   tile (nt, kk): lane l holds 16 contiguous K at k = kk*32 + 16*(l>>4) + j,
//   column n = nt*16 + (l&15).  Stored as ((nt*KK + kk)*32 + lane)*16 + j.
__global__ void prep_weights(const float* __restrict__ Wx0, const float* __restrict__ Wh0,
                             const float* __restrict__ Wd0,
                             const float* __restrict__ Wx1, const float* __restrict__ Wh1,
                             const float* __restrict__ Wd1,
                             unsigned short* __restrict__ wcat0, unsigned short* __restrict__ wd0s,
                             unsigned short* __restrict__ wcat1, unsigned short* __restrict__ wd1s)
{
  const int total = 2 * (WCAT_ELEMS + WD_ELEMS);
  for (int e = blockIdx.x * blockDim.x + threadIdx.x; e < total;
       e += gridDim.x * blockDim.x) {
    int idx = e;
    const float* Wx = nullptr; const float* Wh = nullptr; const float* Wd = nullptr;
    unsigned short* dst; bool isCat;
    if (idx < WCAT_ELEMS)                          { Wx = Wx0; Wh = Wh0; dst = wcat0; isCat = true; }
    else if (idx < WCAT_ELEMS + WD_ELEMS)          { idx -= WCAT_ELEMS; Wd = Wd0; dst = wd0s; isCat = false; }
    else if (idx < 2 * WCAT_ELEMS + WD_ELEMS)      { idx -= WCAT_ELEMS + WD_ELEMS; Wx = Wx1; Wh = Wh1; dst = wcat1; isCat = true; }
    else                                           { idx -= 2 * WCAT_ELEMS + WD_ELEMS; Wd = Wd1; dst = wd1s; isCat = false; }

    const int j    = idx & 15;
    const int lane = (idx >> 4) & 31;
    const int t    = idx >> 9;
    float val;
    if (isCat) {
      const int kk = t & 15;                 // 16 k-steps (K=512)
      const int nt = t >> 4;                 // 64 n-tiles (N=1024)
      const int k  = kk * 32 + ((lane >> 4) << 4) + j;
      const int n  = (nt << 4) + (lane & 15);
      val = (k < 256) ? Wx[k * 1024 + n] : Wh[(k - 256) * 1024 + n];
    } else {
      const int kk = t & 7;                  // 8 k-steps (K=256)
      const int nt = t >> 3;                 // 16 n-tiles
      const int k  = kk * 32 + ((lane >> 4) << 4) + j;
      const int n  = (nt << 4) + (lane & 15);
      val = Wd[k * 256 + n];
    }
    dst[idx] = f2bf(val);
  }
}

// B fragment: 16 contiguous bf16 per lane (two 16B loads)
__device__ __forceinline__ void ldfrag_b(BF16Frag& f, const unsigned short* p) {
  f.u[0] = *reinterpret_cast<const uint4*>(p);
  f.u[1] = *reinterpret_cast<const uint4*>(p + 8);
}
// A fragment: 16 contiguous K per lane (matches B swizzle).
__device__ __forceinline__ void ldfrag_a(BF16Frag& f, const unsigned short* p) {
  f.u[0] = *reinterpret_cast<const uint4*>(p);
  f.u[1] = *reinterpret_cast<const uint4*>(p + 8);
}

template <bool FIRST>
__global__ __launch_bounds__(512)
void tlstm_scan(const float* __restrict__ xin,            // FIRST: layer input fp32
                const unsigned short* __restrict__ yin,   // !FIRST: layer input bf16
                const unsigned short* __restrict__ wcat,  // swizzled [Wx;Wh] bf16
                const unsigned short* __restrict__ wds,   // swizzled Wd bf16
                const float* __restrict__ b,              // [4H]
                const float* __restrict__ bd,             // [H]
                unsigned short* __restrict__ ybf,         // FIRST: y out bf16 (ws)
                float* __restrict__ yf,                   // !FIRST: y out fp32 (d_out)
                float* __restrict__ outH, float* __restrict__ outC)
{
  __shared__ unsigned short Abuf[16][512 + 8];   // [m][k]: k<256 = x_t, k>=256 = h
  __shared__ unsigned short Cbuf[16][256 + 8];   // [m][k]: c state (bf16 copy)
  __shared__ float decs[16];

  const int tid  = threadIdx.x;
  const int lane = tid & 31;
  const int w    = tid >> 5;           // wave 0..15: owns gate columns [16w,16w+16)
  const int half = lane >> 4;
  const int nl   = lane & 15;
  const int col  = (w << 4) + nl;      // hidden column 0..255
  const int b0   = blockIdx.x << 4;    // batch tile base
  const int laneoff = lane * 16;       // per-lane element offset inside a frag
  const size_t wdoff   = (size_t)w * 4096 + laneoff;   // nt = w tile in Wd
  const size_t wcatoff = (size_t)w * 8192 + laneoff;   // nt = w (+16g) in Wcat

  // per-lane biases (constant over the scan)
  const float bi  = b[0 * Hdim + col];
  const float bfg = b[1 * Hdim + col];
  const float bo  = b[2 * Hdim + col];
  const float bc  = b[3 * Hdim + col];
  const float bdv = bd[col];

  // zero the h half of A (h_{-1} = 0)
  for (int e = tid; e < 16 * 256; e += 512) Abuf[e >> 8][256 + (e & 255)] = 0;

  v8f creg = {0.f, 0.f, 0.f, 0.f, 0.f, 0.f, 0.f, 0.f};   // c state (C-layout)
  v8f hN   = {0.f, 0.f, 0.f, 0.f, 0.f, 0.f, 0.f, 0.f};

  for (int s = 0; s < Sdim; ++s) {
    // ---- stage x_t into A[:, 0:256] (bf16) ----
    if (FIRST) {
      for (int e = tid; e < 16 * 256; e += 512) {
        const int m = e >> 8, k = e & 255;
        // read-once stream: non-temporal so it doesn't evict the weight set
        const float v = __builtin_nontemporal_load(
            &xin[(((size_t)(b0 + m)) * Sdim + s) * Hdim + k]);
        Abuf[m][k] = f2bf(v);
      }
    } else {
      for (int e = tid; e < 16 * 256; e += 512) {
        const int m = e >> 8, k = e & 255;
        Abuf[m][k] = yin[(((size_t)(b0 + m)) * Sdim + s) * Hdim + k];
      }
    }
    // ---- per-row time decay: 1/log(e + x_t[..., -1]) ----
    if (tid < 16) {
      float dt;
      if (FIRST) dt = xin[(((size_t)(b0 + tid)) * Sdim + s) * Hdim + (Hdim - 1)];
      else       dt = bf2f(yin[(((size_t)(b0 + tid)) * Sdim + s) * Hdim + (Hdim - 1)]);
      decs[tid] = 1.0f / logf(2.718281828459045f + dt);
    }
    // ---- publish c (bf16) for the c@Wd GEMM ----
#pragma unroll
    for (int r = 0; r < 8; ++r) {
      const int m = r + (half << 3);
      Cbuf[m][col] = f2bf(creg[r]);
    }
    __syncthreads();

    // ---- cs = tanh(c @ Wd + bd), this wave's 16 columns ----
    v8f accs = {0.f, 0.f, 0.f, 0.f, 0.f, 0.f, 0.f, 0.f};
#pragma unroll
    for (int kk = 0; kk < 8; ++kk) {
      BF16Frag a, fb;
      ldfrag_a(a, &Cbuf[nl][kk * 32 + (half << 4)]);
      ldfrag_b(fb, wds + wdoff + lau(kk * 512));
      accs = __builtin_amdgcn_wmma_f32_16x16x32_bf16(false, a.v, false, fb.v,
                                                     (short)0, accs, false, false);
    }

    // ---- g = [x_t, h] @ [Wx; Wh] + b : four gate tiles, K = 512 ----
    v8f ai = {bi, bi, bi, bi, bi, bi, bi, bi};
    v8f af = {bfg, bfg, bfg, bfg, bfg, bfg, bfg, bfg};
    v8f ao = {bo, bo, bo, bo, bo, bo, bo, bo};
    v8f ac = {bc, bc, bc, bc, bc, bc, bc, bc};
#pragma unroll
    for (int kk = 0; kk < 16; ++kk) {
      BF16Frag a;
      ldfrag_a(a, &Abuf[nl][kk * 32 + (half << 4)]);
      const unsigned short* wp = wcat + wcatoff + lau(kk * 512);
      BF16Frag fb;
      ldfrag_b(fb, wp);                                // gate i    (nt = w)
      ai = __builtin_amdgcn_wmma_f32_16x16x32_bf16(false, a.v, false, fb.v, (short)0, ai, false, false);
      ldfrag_b(fb, wp + 131072);                       // gate f    (nt = w+16)
      af = __builtin_amdgcn_wmma_f32_16x16x32_bf16(false, a.v, false, fb.v, (short)0, af, false, false);
      ldfrag_b(fb, wp + 262144);                       // gate o    (nt = w+32)
      ao = __builtin_amdgcn_wmma_f32_16x16x32_bf16(false, a.v, false, fb.v, (short)0, ao, false, false);
      ldfrag_b(fb, wp + 393216);                       // gate cand (nt = w+48)
      ac = __builtin_amdgcn_wmma_f32_16x16x32_bf16(false, a.v, false, fb.v, (short)0, ac, false, false);
    }

    // ---- elementwise cell update (fp32, identical C-layout lanes) ----
#pragma unroll
    for (int r = 0; r < 8; ++r) {
      const float dm = decs[r + (half << 3)];
      const float cs = tanhf(accs[r] + bdv);
      const float ca = creg[r] - cs + cs * dm;
      const float ig = sigf(ai[r]);
      const float fg = sigf(af[r]);
      const float og = sigf(ao[r]);
      const float cd = tanhf(ac[r]);
      const float cn = fg * ca + ig * cd;
      creg[r] = cn;
      hN[r]   = og * tanhf(cn);
    }

    __syncthreads();   // everyone done reading A's h half before we overwrite it

    // ---- publish h_new for the next step + emit this step's output ----
#pragma unroll
    for (int r = 0; r < 8; ++r) {
      const int m = r + (half << 3);
      const unsigned short hb = f2bf(hN[r]);
      Abuf[m][256 + col] = hb;
      const size_t oidx = (((size_t)(b0 + m)) * Sdim + s) * Hdim + col;
      if (FIRST) ybf[oidx] = hb;
      else       yf[oidx]  = hN[r];
    }
    // next iteration's first barrier orders these writes before the GEMM reads
  }

  // ---- final states ----
#pragma unroll
  for (int r = 0; r < 8; ++r) {
    const int m = r + (half << 3);
    outH[(size_t)(b0 + m) * Hdim + col] = hN[r];
    outC[(size_t)(b0 + m) * Hdim + col] = creg[r];
  }
}

extern "C" void kernel_launch(void* const* d_in, const int* in_sizes, int n_in,
                              void* d_out, int out_size, void* d_ws, size_t ws_size,
                              hipStream_t stream) {
  (void)in_sizes; (void)n_in; (void)out_size; (void)ws_size;
  const float* x   = (const float*)d_in[0];
  const float* Wx0 = (const float*)d_in[1];
  const float* Wh0 = (const float*)d_in[2];
  const float* b0p = (const float*)d_in[3];
  const float* Wd0 = (const float*)d_in[4];
  const float* bd0 = (const float*)d_in[5];
  const float* Wx1 = (const float*)d_in[6];
  const float* Wh1 = (const float*)d_in[7];
  const float* b1p = (const float*)d_in[8];
  const float* Wd1 = (const float*)d_in[9];
  const float* bd1 = (const float*)d_in[10];

  // workspace layout (bf16): wcat0 | wd0 | wcat1 | wd1 | y0   (~70 MB)
  unsigned short* wsp   = (unsigned short*)d_ws;
  unsigned short* wcat0 = wsp;
  unsigned short* wd0s  = wcat0 + WCAT_ELEMS;
  unsigned short* wcat1 = wd0s + WD_ELEMS;
  unsigned short* wd1s  = wcat1 + WCAT_ELEMS;
  unsigned short* y0bf  = wd1s + WD_ELEMS;

  float* out = (float*)d_out;
  float* y1  = out;                                  // [B,S,H]
  float* h0  = out + (size_t)Bdim * Sdim * Hdim;     // [B,H]
  float* c0  = h0 + (size_t)Bdim * Hdim;
  float* h1  = c0 + (size_t)Bdim * Hdim;
  float* c1  = h1 + (size_t)Bdim * Hdim;

  prep_weights<<<2304, 512, 0, stream>>>(Wx0, Wh0, Wd0, Wx1, Wh1, Wd1,
                                         wcat0, wd0s, wcat1, wd1s);

  tlstm_scan<true><<<Bdim / 16, 512, 0, stream>>>(
      x, nullptr, wcat0, wd0s, b0p, bd0, y0bf, nullptr, h0, c0);

  tlstm_scan<false><<<Bdim / 16, 512, 0, stream>>>(
      nullptr, y0bf, wcat1, wd1s, b1p, bd1, nullptr, y1, h1, c1);
}
